// EEGGAT_68599217651783
// MI455X (gfx1250) — compile-verified, matrix operand order checked by reading, not run on hardware
//
#include <hip/hip_runtime.h>
#include <hip/hip_bf16.h>
#include <math.h>

// ---------------------------------------------------------------------------
// GAT 3-layer pipeline for MI455X (gfx1250), bf16 WMMA GEMMs + atomic scatter
// ---------------------------------------------------------------------------

typedef __bf16 bf16;
typedef __attribute__((ext_vector_type(16))) __bf16 v16bf;
typedef __attribute__((ext_vector_type(8)))  float  v8f;

constexpr int kN   = 30000;
constexpr int kFin = 3000;
constexpr int kH   = 4;
constexpr int kC   = 64;
constexpr int kHC  = 256;
constexpr int kE   = 570000;
constexpr int kET  = 600000;          // edges + self loops
constexpr int kG   = 1500;
constexpr int kMT  = kN / 16;         // 1875 M tiles
constexpr int kNT  = kHC / 16;        // 16 N tiles
constexpr int kKT1 = (kFin + 31)/32;  // 94 K tiles (padded 3000->3008)
constexpr int kKT2 = kHC / 32;        // 8 K tiles

// ---- workspace layout (bytes) ----
constexpr size_t ALGN(size_t x){ return (x + 255) & ~(size_t)255; }
constexpr size_t SZ_APACK = (size_t)kMT * kKT1 * 512 * 2;   // bf16 A fragments
constexpr size_t SZ_WPACK = (size_t)kKT1 * kNT * 512 * 2;   // bf16 B fragments
constexpr size_t SZ_NHC   = (size_t)kN * kHC * 4;
constexpr size_t SZ_NH    = (size_t)kN * kH * 4;
constexpr size_t OFF_APACK = 0;
constexpr size_t OFF_WPACK = ALGN(OFF_APACK + SZ_APACK);
constexpr size_t OFF_HBUF  = ALGN(OFF_WPACK + SZ_WPACK);    // x@W result  [N,HC] f32
constexpr size_t OFF_OUT   = ALGN(OFF_HBUF  + SZ_NHC);      // attn output [N,HC] f32
constexpr size_t OFF_CUR   = ALGN(OFF_OUT   + SZ_NHC);      // layer input [N,HC] f32
constexpr size_t OFF_ASN   = ALGN(OFF_CUR   + SZ_NHC);      // [N,H]
constexpr size_t OFF_ADN   = ALGN(OFF_ASN   + SZ_NH);
constexpr size_t OFF_EMAX  = ALGN(OFF_ADN   + SZ_NH);       // uint-encoded max
constexpr size_t OFF_DEN   = ALGN(OFF_EMAX  + SZ_NH);
constexpr size_t OFF_EBUF  = ALGN(OFF_DEN   + SZ_NH);       // [ET,H]
constexpr size_t OFF_CNT   = ALGN(OFF_EBUF  + (size_t)kET * kH * 4);

// ---- helpers ----
__device__ inline bf16 f2bf(float f) {
  unsigned u = __float_as_uint(f);
  unsigned r = (u + 0x7FFFu + ((u >> 16) & 1u)) >> 16;   // round-to-nearest-even
  unsigned short s = (unsigned short)r;
  return *reinterpret_cast<bf16*>(&s);
}
__device__ inline unsigned fenc(float f) {               // monotonic uint encoding
  unsigned u = __float_as_uint(f);
  return (u & 0x80000000u) ? ~u : (u | 0x80000000u);
}
__device__ inline float fdec(unsigned u) {
  return __uint_as_float((u & 0x80000000u) ? (u ^ 0x80000000u) : ~u);
}
constexpr unsigned ENC_NEGINF = 0x007FFFFFu;             // fenc(-inf)

__device__ inline void edge_ep(const int* __restrict__ ei, int e, int& s, int& d) {
  if (e < kE) { s = ei[e]; d = ei[kE + e]; } else { s = d = e - kE; }
}

// ---------------------------------------------------------------------------
// Repack row-major f32 A [M,K] into per-wave WMMA A fragments (16x32 bf16)
// frag element j of lane l:  m = mt*16 + (l&15),
//                            k = kt*32 + (l>>4)*8 + (j>=8?16:0) + (j&7)
// ---------------------------------------------------------------------------
__global__ void repackA_kernel(const float* __restrict__ src, bf16* __restrict__ dst,
                               int mtiles, int ktiles, int K, int lda) {
  int tile = blockIdx.x * (blockDim.x >> 5) + (threadIdx.x >> 5);
  int lane = threadIdx.x & 31;
  if (tile >= mtiles * ktiles) return;
  int mt = tile / ktiles, kt = tile % ktiles;
  int m = mt * 16 + (lane & 15);
  int kbase = kt * 32 + ((lane >> 4) << 3);
  bf16* d = dst + (size_t)tile * 512 + lane * 16;
  const float* s = src + (size_t)m * lda;
#pragma unroll
  for (int j = 0; j < 16; ++j) {
    int k = kbase + ((j >> 3) << 4) + (j & 7);
    d[j] = f2bf(k < K ? s[k] : 0.f);
  }
}

// Repack row-major f32 W [K,N=256] into WMMA B fragments (32x16 bf16)
__global__ void repackB_kernel(const float* __restrict__ src, bf16* __restrict__ dst,
                               int ktiles, int K) {
  int tile = blockIdx.x * (blockDim.x >> 5) + (threadIdx.x >> 5);
  int lane = threadIdx.x & 31;
  if (tile >= ktiles * kNT) return;
  int kt = tile / kNT, nt = tile % kNT;
  int n = nt * 16 + (lane & 15);
  int kbase = kt * 32 + ((lane >> 4) << 3);
  bf16* d = dst + (size_t)tile * 512 + lane * 16;
#pragma unroll
  for (int j = 0; j < 16; ++j) {
    int k = kbase + ((j >> 3) << 4) + (j & 7);
    d[j] = f2bf(k < K ? src[(size_t)k * kHC + n] : 0.f);
  }
}

// ---------------------------------------------------------------------------
// bf16 WMMA GEMM:  out[N,HC] f32 = Apack x Wpack.
// Each wave computes a 16x128 strip: 1 A-frag load feeds 8 WMMAs per K-step,
// halving A-stream traffic vs 16x64 (A is the only non-L2-resident operand).
// ---------------------------------------------------------------------------
__global__ __launch_bounds__(256) void gemm_wmma_kernel(
    const bf16* __restrict__ apack, const bf16* __restrict__ bpack,
    float* __restrict__ out, int mtiles, int ktiles) {
  int wave = blockIdx.x * (blockDim.x >> 5) + (threadIdx.x >> 5);
  int lane = threadIdx.x & 31;
  int mt = wave >> 1;          // 2 N-groups of 128 columns
  int ng = wave & 1;
  if (mt >= mtiles) return;    // wave-uniform: EXEC stays all-ones below

  v8f acc[8];
#pragma unroll
  for (int j = 0; j < 8; ++j) acc[j] = (v8f){};

  const bf16* ap = apack + (size_t)mt * ktiles * 512 + lane * 16;
  const bf16* bp = bpack + (size_t)(ng * 8) * 512 + lane * 16;

  for (int kt = 0; kt < ktiles; ++kt) {
    __builtin_prefetch(ap + 512, 0, 3);                  // near-cache prefetch
    v16bf a = *(const v16bf*)ap; ap += 512;
    const bf16* b = bp + (size_t)kt * kNT * 512;
#pragma unroll
    for (int j = 0; j < 8; ++j) {
      v16bf bj = *(const v16bf*)(b + (size_t)j * 512);
      acc[j] = __builtin_amdgcn_wmma_f32_16x16x32_bf16(false, a, false, bj,
                                                       (short)0, acc[j], false, false);
    }
  }

  // C/D layout: VGPR r -> row (lane>=16 ? 8 : 0)+r, col = lane&15
  int row = mt * 16 + ((lane >> 4) << 3);
  int col = ng * 128 + (lane & 15);
  float* o = out + (size_t)row * kHC + col;
#pragma unroll
  for (int j = 0; j < 8; ++j)
#pragma unroll
    for (int r = 0; r < 8; ++r)
      o[(size_t)r * kHC + j * 16] = acc[j][r];
}

// ---------------------------------------------------------------------------
// Attention pipeline
// ---------------------------------------------------------------------------
__global__ void attn_prep_kernel(const float* __restrict__ hbuf,
                                 const float* __restrict__ asrc,
                                 const float* __restrict__ adst,
                                 float* __restrict__ asn, float* __restrict__ adn,
                                 unsigned* __restrict__ emax, float* __restrict__ denom) {
  int idx = blockIdx.x * blockDim.x + threadIdx.x;
  if (idx >= kN * kH) return;
  int n = idx >> 2, h = idx & 3;
  const float* hp = hbuf + (size_t)n * kHC + h * kC;
  float sa = 0.f, sd = 0.f;
#pragma unroll 8
  for (int c = 0; c < kC; ++c) {
    float v = hp[c];
    sa += v * asrc[h * kC + c];
    sd += v * adst[h * kC + c];
  }
  asn[idx] = sa; adn[idx] = sd;
  emax[idx] = ENC_NEGINF; denom[idx] = 0.f;
}

__global__ void zero_kernel(float* __restrict__ p, int n) {
  int idx = blockIdx.x * blockDim.x + threadIdx.x;
  if (idx < n) p[idx] = 0.f;
}

__global__ void edge_max_kernel(const int* __restrict__ ei,
                                const float* __restrict__ asn, const float* __restrict__ adn,
                                float* __restrict__ ebuf, unsigned* __restrict__ emax) {
  int idx = blockIdx.x * blockDim.x + threadIdx.x;
  if (idx >= kET * kH) return;
  int e = idx >> 2, h = idx & 3;
  int s, d; edge_ep(ei, e, s, d);
  float v = asn[s * kH + h] + adn[d * kH + h];
  v = v > 0.f ? v : 0.2f * v;                 // leaky relu
  ebuf[idx] = v;
  atomicMax(&emax[d * kH + h], fenc(v));
}

__global__ void edge_sum_kernel(const int* __restrict__ ei,
                                float* __restrict__ ebuf,
                                const unsigned* __restrict__ emax,
                                float* __restrict__ denom) {
  int idx = blockIdx.x * blockDim.x + threadIdx.x;
  if (idx >= kET * kH) return;
  int e = idx >> 2, h = idx & 3;
  int s, d; edge_ep(ei, e, s, d);
  float ee = __expf(ebuf[idx] - fdec(emax[d * kH + h]));
  ebuf[idx] = ee;
  atomicAdd(&denom[d * kH + h], ee);
}

__global__ __launch_bounds__(256) void edge_scatter_kernel(
    const int* __restrict__ ei, const float* __restrict__ hbuf,
    const float* __restrict__ ebuf, const float* __restrict__ denom,
    float* __restrict__ out) {
  int e = blockIdx.x;
  int t = threadIdx.x;          // hc index 0..255
  int s, d; edge_ep(ei, e, s, d);
  int h = t >> 6;
  float alpha = ebuf[(size_t)e * kH + h] / (denom[d * kH + h] + 1e-16f);
  atomicAdd(&out[(size_t)d * kHC + t], hbuf[(size_t)s * kHC + t] * alpha);
}

__global__ void bias_bn_elu_kernel(const float* __restrict__ src,
                                   const float* __restrict__ bias,
                                   const float* __restrict__ g, const float* __restrict__ bb,
                                   const float* __restrict__ m, const float* __restrict__ v,
                                   float* __restrict__ dst) {
  int idx = blockIdx.x * blockDim.x + threadIdx.x;
  if (idx >= kN * kHC) return;
  int f = idx & (kHC - 1);
  float x = src[idx] + bias[f];
  float y = (x - m[f]) * rsqrtf(v[f] + 1e-5f) * g[f] + bb[f];
  dst[idx] = y > 0.f ? y : __expf(y) - 1.f;
}

// layer 3: mean over heads + bias + BN + ELU + graph-sum scatter
__global__ void head_mean_scatter_kernel(const float* __restrict__ src,
                                         const float* __restrict__ bias,
                                         const float* __restrict__ g, const float* __restrict__ bb,
                                         const float* __restrict__ m, const float* __restrict__ v,
                                         const int* __restrict__ batch,
                                         float* __restrict__ gout) {
  int idx = blockIdx.x * blockDim.x + threadIdx.x;
  if (idx >= kN * kC) return;
  int n = idx >> 6, c = idx & 63;
  const float* o = src + (size_t)n * kHC;
  float x = 0.25f * (o[c] + o[64 + c] + o[128 + c] + o[192 + c]) + bias[c];
  float y = (x - m[c]) * rsqrtf(v[c] + 1e-5f) * g[c] + bb[c];
  y = y > 0.f ? y : __expf(y) - 1.f;
  atomicAdd(&gout[(size_t)batch[n] * kC + c], y);
}

__global__ void count_kernel(const int* __restrict__ batch, float* __restrict__ cnt) {
  int n = blockIdx.x * blockDim.x + threadIdx.x;
  if (n < kN) atomicAdd(&cnt[batch[n]], 1.f);
}

__global__ void divide_kernel(float* __restrict__ gout, const float* __restrict__ cnt) {
  int idx = blockIdx.x * blockDim.x + threadIdx.x;
  if (idx >= kG * kC) return;
  gout[idx] /= fmaxf(cnt[idx >> 6], 1.f);
}

// ---------------------------------------------------------------------------
extern "C" void kernel_launch(void* const* d_in, const int* in_sizes, int n_in,
                              void* d_out, int out_size, void* d_ws, size_t ws_size,
                              hipStream_t stream) {
  const float* x     = (const float*)d_in[0];
  const int*   ei    = (const int*)d_in[1];
  const int*   batch = (const int*)d_in[2];
  const float* Wm[3] = {(const float*)d_in[3],  (const float*)d_in[11], (const float*)d_in[19]};
  const float* AS[3] = {(const float*)d_in[4],  (const float*)d_in[12], (const float*)d_in[20]};
  const float* AD[3] = {(const float*)d_in[5],  (const float*)d_in[13], (const float*)d_in[21]};
  const float* BI[3] = {(const float*)d_in[6],  (const float*)d_in[14], (const float*)d_in[22]};
  const float* BG[3] = {(const float*)d_in[7],  (const float*)d_in[15], (const float*)d_in[23]};
  const float* BB[3] = {(const float*)d_in[8],  (const float*)d_in[16], (const float*)d_in[24]};
  const float* BM[3] = {(const float*)d_in[9],  (const float*)d_in[17], (const float*)d_in[25]};
  const float* BV[3] = {(const float*)d_in[10], (const float*)d_in[18], (const float*)d_in[26]};

  char* ws = (char*)d_ws;
  bf16*     apack = (bf16*)(ws + OFF_APACK);
  bf16*     wpack = (bf16*)(ws + OFF_WPACK);
  float*    hbuf  = (float*)(ws + OFF_HBUF);
  float*    outb  = (float*)(ws + OFF_OUT);
  float*    cur   = (float*)(ws + OFF_CUR);
  float*    asn   = (float*)(ws + OFF_ASN);
  float*    adn   = (float*)(ws + OFF_ADN);
  unsigned* emax  = (unsigned*)(ws + OFF_EMAX);
  float*    denom = (float*)(ws + OFF_DEN);
  float*    ebuf  = (float*)(ws + OFF_EBUF);
  float*    cnt   = (float*)(ws + OFF_CNT);
  float*    gout  = (float*)d_out;

  const int TPB = 256;
  for (int l = 0; l < 3; ++l) {
    const float* inp = (l == 0) ? x : cur;
    int K      = (l == 0) ? kFin : kHC;
    int ktiles = (l == 0) ? kKT1 : kKT2;

    int atiles = kMT * ktiles;
    repackA_kernel<<<(atiles + 7) / 8, TPB, 0, stream>>>(inp, apack, kMT, ktiles, K, K);
    int btiles = ktiles * kNT;
    repackB_kernel<<<(btiles + 7) / 8, TPB, 0, stream>>>(Wm[l], wpack, ktiles, K);

    int waves = kMT * 2;
    gemm_wmma_kernel<<<(waves + 7) / 8, TPB, 0, stream>>>(apack, wpack, hbuf, kMT, ktiles);

    attn_prep_kernel<<<(kN * kH + TPB - 1) / TPB, TPB, 0, stream>>>(hbuf, AS[l], AD[l],
                                                                    asn, adn, emax, denom);
    zero_kernel<<<(kN * kHC + TPB - 1) / TPB, TPB, 0, stream>>>(outb, kN * kHC);
    edge_max_kernel<<<(kET * kH + TPB - 1) / TPB, TPB, 0, stream>>>(ei, asn, adn, ebuf, emax);
    edge_sum_kernel<<<(kET * kH + TPB - 1) / TPB, TPB, 0, stream>>>(ei, ebuf, emax, denom);
    edge_scatter_kernel<<<kET, TPB, 0, stream>>>(ei, hbuf, ebuf, denom, outb);

    if (l < 2) {
      bias_bn_elu_kernel<<<(kN * kHC + TPB - 1) / TPB, TPB, 0, stream>>>(
          outb, BI[l], BG[l], BB[l], BM[l], BV[l], cur);
    }
  }

  // final: head mean + BN + ELU + graph mean
  zero_kernel<<<(kG * kC + TPB - 1) / TPB, TPB, 0, stream>>>(gout, kG * kC);
  zero_kernel<<<(kG + TPB - 1) / TPB, TPB, 0, stream>>>(cnt, kG);
  head_mean_scatter_kernel<<<(kN * kC + TPB - 1) / TPB, TPB, 0, stream>>>(
      outb, BI[2], BG[2], BB[2], BM[2], BV[2], batch, gout);
  count_kernel<<<(kN + TPB - 1) / TPB, TPB, 0, stream>>>(batch, cnt);
  divide_kernel<<<(kG * kC + TPB - 1) / TPB, TPB, 0, stream>>>(gout, cnt);
}
